// DuplicationRemovalNetwork_39204461478680
// MI455X (gfx1250) — matrix operand order repaired and checked by reading, not compiled, not run on hardware
//
#include <hip/hip_runtime.h>
#include <hip/hip_bf16.h>

#define NPROP 1024
#define CCLS  2
#define FEATD 1024
#define GEO   80
#define GROUP 16
#define DG    64
#define TOPK  20
#define LOG_EPS  (-13.815511f)
#define EPSF     (1e-6f)
#define NEG_INF  (-3.0e38f)

typedef __bf16 bf16;
typedef bf16  v16bf __attribute__((ext_vector_type(16)));
typedef bf16  v8bf  __attribute__((ext_vector_type(8)));
typedef float v8f   __attribute__((ext_vector_type(8)));

// ---------------------------------------------------------------------------
// WMMA helpers (CDNA5 wave32, 16x16x32 bf16 -> f32)
// ---------------------------------------------------------------------------
__device__ __forceinline__ v8f wmma_bf16(v16bf a, v16bf b, v8f c) {
  return __builtin_amdgcn_wmma_f32_16x16x32_bf16(false, a, false, b, (short)0, c,
                                                 false, false);
}

// A fragment (16x32, row-major, k contiguous). Lane L: row = L&15, h = L>>4,
// k elements [h*8, h*8+8) and [h*8+16, h*8+24)  (per ISA 16-bit A layout).
__device__ __forceinline__ v16bf load_a_frag(const bf16* tile, int ld, int lane) {
  int r = lane & 15, h = lane >> 4;
  const bf16* p = tile + (size_t)r * ld + h * 8;
  v8bf lo = *(const v8bf*)(p);
  v8bf hi = *(const v8bf*)(p + 16);
  v16bf o;
#pragma unroll
  for (int i = 0; i < 8; ++i) { o[i] = lo[i]; o[8 + i] = hi[i]; }
  return o;
}

// B fragment (32x16) loaded from a K-contiguous transposed tile BT[n][k].
// Lane L: column n = L&15, h = L>>4, k elements [h*8, h*8+8) and [h*8+16, h*8+24).
__device__ __forceinline__ v16bf load_b_frag(const bf16* tileT, int ld, int lane) {
  int n = lane & 15, h = lane >> 4;
  const bf16* p = tileT + (size_t)n * ld + h * 8;
  v8bf lo = *(const v8bf*)(p);
  v8bf hi = *(const v8bf*)(p + 16);
  v16bf o;
#pragma unroll
  for (int i = 0; i < 8; ++i) { o[i] = lo[i]; o[8 + i] = hi[i]; }
  return o;
}

// ---------------------------------------------------------------------------
// K0a: f_a [N,C,F] f32 -> fa_bf [C,N,F] bf16
// ---------------------------------------------------------------------------
__global__ void k_convert_A(const float* __restrict__ fa, bf16* __restrict__ out) {
  int tid = blockIdx.x * 256 + threadIdx.x;      // 0 .. 2M-1
  int f = tid & 1023;
  int row = tid >> 10;                           // c*N + n
  int c = row >> 10, n = row & 1023;
  out[tid] = (bf16)fa[((size_t)n * CCLS + c) * FEATD + f];
}

// K0b: W [F, GD] f32 -> WT [GD, F] bf16 (k-contiguous for B fragments)
__global__ void k_transpose_w(const float* __restrict__ w, bf16* __restrict__ out) {
  int tid = blockIdx.x * 256 + threadIdx.x;      // 0 .. 1M-1
  int gd = tid >> 10, f = tid & 1023;
  out[tid] = (bf16)w[(size_t)f * 1024 + gd];
}

// K0c: conv_w [G*DG, F] is already k(=f)-contiguous; just convert to bf16.
__global__ void k_convert_w(const float* __restrict__ w, bf16* __restrict__ out) {
  int tid = blockIdx.x * 256 + threadIdx.x;
  out[tid] = (bf16)w[tid];
}

// ---------------------------------------------------------------------------
// K1: fused gate = log(clamp(relu(pos@WG + b), eps)) + log_iou
//     one streaming pass over the 671 MB position_embedding tensor.
//     output layout [C,G,N,N] bf16 (unit-stride in m for the attention kernel)
// ---------------------------------------------------------------------------
__global__ void k_gate(const float* __restrict__ pos, const float* __restrict__ iou,
                       const float* __restrict__ WGw, const float* __restrict__ WGb,
                       bf16* __restrict__ gate) {
  __shared__ float wg[GEO * GROUP];
  __shared__ float wb[GROUP];
  int tid = threadIdx.x;
  for (int i = tid; i < GEO * GROUP; i += 256) wg[i] = WGw[i];
  if (tid < GROUP) wb[tid] = WGb[tid];
  __syncthreads();

  int e = blockIdx.x * 256 + tid;                // (c,n,m) flat, 0 .. 2M-1
  int m = e & 1023, n = (e >> 10) & 1023, c = e >> 20;

  const float4* pv4 = (const float4*)(pos + (size_t)e * GEO);
  float acc[GROUP];
#pragma unroll
  for (int g = 0; g < GROUP; ++g) acc[g] = wb[g];
  for (int q = 0; q < GEO / 4; ++q) {
    float4 p4 = pv4[q];
    int k = q * 4;
#pragma unroll
    for (int g = 0; g < GROUP; ++g)
      acc[g] += p4.x * wg[k * GROUP + g] + p4.y * wg[(k + 1) * GROUP + g] +
                p4.z * wg[(k + 2) * GROUP + g] + p4.w * wg[(k + 3) * GROUP + g];
  }
  float liou = (iou[e] >= EPSF) ? 0.f : LOG_EPS;
#pragma unroll
  for (int g = 0; g < GROUP; ++g) {
    float x = acc[g];                            // relu + clamp folded:
    float gl = (x >= EPSF) ? __logf(x) : LOG_EPS;
    gate[((size_t)((c * GROUP + g) * NPROP) + n) * NPROP + m] = (bf16)(gl + liou);
  }
}

// ---------------------------------------------------------------------------
// K2: WMMA GEMM  D[2048,1024] = A[2048,1024] x B[1024,1024] (+bias)
//     A bf16 row-major, BT bf16 [col][k]; B (2 MB) stays resident in L2.
//     block = 256 threads (8 waves); wave -> 16 rows x 64 cols.
//     Software-pipelined: fragments for k-step k0+32 are loaded before the
//     WMMAs of step k0 issue, so waits become partial (no loadcnt==0 stalls).
// ---------------------------------------------------------------------------
__global__ void k_gemm(const bf16* __restrict__ A, const bf16* __restrict__ BT,
                       const float* __restrict__ bias,
                       bf16* __restrict__ outB, float* __restrict__ outF) {
  int lane = threadIdx.x & 31, wave = threadIdx.x >> 5;
  int row0 = blockIdx.x * 16;
  int colBase = blockIdx.y * 512 + wave * 64;

  const bf16* Abase = A + (size_t)row0 * 1024;

  v8f acc[4];
#pragma unroll
  for (int t = 0; t < 4; ++t) acc[t] = (v8f){0.f, 0.f, 0.f, 0.f, 0.f, 0.f, 0.f, 0.f};

  // prologue: fragments for k0 = 0
  v16bf a_cur = load_a_frag(Abase, 1024, lane);
  v16bf b_cur[4];
#pragma unroll
  for (int t = 0; t < 4; ++t)
    b_cur[t] = load_b_frag(BT + (size_t)(colBase + t * 16) * 1024, 1024, lane);

  // steady state: prefetch k0+32, then WMMA on k0
  for (int k0 = 0; k0 < 1024 - 32; k0 += 32) {
    int kn = k0 + 32;
    v16bf a_nxt = load_a_frag(Abase + kn, 1024, lane);
    v16bf b_nxt[4];
#pragma unroll
    for (int t = 0; t < 4; ++t)
      b_nxt[t] = load_b_frag(BT + (size_t)(colBase + t * 16) * 1024 + kn, 1024, lane);
#pragma unroll
    for (int t = 0; t < 4; ++t) acc[t] = wmma_bf16(a_cur, b_cur[t], acc[t]);
    a_cur = a_nxt;
#pragma unroll
    for (int t = 0; t < 4; ++t) b_cur[t] = b_nxt[t];
  }
  // epilogue: last k-step
#pragma unroll
  for (int t = 0; t < 4; ++t) acc[t] = wmma_bf16(a_cur, b_cur[t], acc[t]);

  int cl = lane & 15, h = lane >> 4;
#pragma unroll
  for (int t = 0; t < 4; ++t) {
    int col = colBase + t * 16 + cl;
    float bv = bias ? bias[col] : 0.f;
#pragma unroll
    for (int i = 0; i < 8; ++i) {
      int r = row0 + i + h * 8;                  // C-layout: VGPR i -> M = i / 8+i
      float v = acc[t][i] + bv;
      if (outF) outF[(size_t)r * 1024 + col] = v;
      else      outB[(size_t)r * 1024 + col] = (bf16)v;
    }
  }
}

// ---------------------------------------------------------------------------
// K3: fused affinity (WMMA QK^T / 8) + gate add + top-20 + softmax.
//     grid (nblk=64, g=16, c=2); block = 128 (4 waves); each block does a
//     16-row x 1024-col score stripe in LDS, then per-row top-k.
//     m-tile loop is double-buffered: next tile's K fragments are in flight
//     while the current tile's WMMAs execute.
// ---------------------------------------------------------------------------
__global__ void k_attn(const bf16* __restrict__ Q, const bf16* __restrict__ K,
                       const bf16* __restrict__ gate,
                       float* __restrict__ topv, int* __restrict__ topi) {
  extern __shared__ char smem[];
  float* sc  = (float*)smem;                     // [16][1024]
  float* tv  = sc + 16 * NPROP;                  // [16][TOPK]
  int*   ti  = (int*)(tv + 16 * TOPK);           // [16][TOPK]
  float* red = (float*)(ti + 16 * TOPK);         // [128]
  int*  redi = (int*)(red + 128);                // [128]

  int tid = threadIdx.x;
  int lane = tid & 31, wave = tid >> 5;
  int c = blockIdx.z, g = blockIdx.y, row0 = blockIdx.x * 16;

  const bf16* Qbase = Q + ((size_t)c * NPROP + row0) * 1024 + g * DG;
  v16bf a0 = load_a_frag(Qbase, 1024, lane);       // d = 0..31
  v16bf a1 = load_a_frag(Qbase + 32, 1024, lane);  // d = 32..63

  // prologue: K fragments for first m-tile of this wave
  const bf16* Kb0 = K + ((size_t)c * NPROP + wave * 256) * 1024 + g * DG;
  v16bf b0 = load_b_frag(Kb0, 1024, lane);
  v16bf b1 = load_b_frag(Kb0 + 32, 1024, lane);

  int cl = lane & 15, h = lane >> 4;
  for (int t = 0; t < 16; ++t) {
    int m0 = wave * 256 + t * 16;
    // prefetch next tile before issuing current WMMAs
    v16bf nb0 = b0, nb1 = b1;
    if (t < 15) {
      const bf16* Kn = K + ((size_t)c * NPROP + m0 + 16) * 1024 + g * DG;
      nb0 = load_b_frag(Kn, 1024, lane);
      nb1 = load_b_frag(Kn + 32, 1024, lane);
    }
    v8f acc = (v8f){0.f, 0.f, 0.f, 0.f, 0.f, 0.f, 0.f, 0.f};
    acc = wmma_bf16(a0, b0, acc);
    acc = wmma_bf16(a1, b1, acc);
    b0 = nb0; b1 = nb1;

    const bf16* gp =
        gate + ((size_t)((c * GROUP + g) * NPROP) + row0) * NPROP + m0 + cl;
#pragma unroll
    for (int i = 0; i < 8; ++i) {
      int r = i + h * 8;
      sc[r * NPROP + m0 + cl] = acc[i] * 0.125f + (float)gp[(size_t)r * NPROP];
    }
  }
  __syncthreads();

  // iterative top-20: 8 threads per row
  int row = tid >> 3, sub = tid & 7;
  for (int j = 0; j < TOPK; ++j) {
    float bm = NEG_INF; int bi = 0;
    for (int m = sub; m < NPROP; m += 8) {
      float v = sc[row * NPROP + m];
      if (v > bm) { bm = v; bi = m; }
    }
    red[tid] = bm; redi[tid] = bi;
    __syncthreads();
    if (sub == 0) {
      for (int s2 = 1; s2 < 8; ++s2)
        if (red[tid + s2] > bm) { bm = red[tid + s2]; bi = redi[tid + s2]; }
      tv[row * TOPK + j] = bm;
      ti[row * TOPK + j] = bi;
      sc[row * NPROP + bi] = NEG_INF;            // mask the winner
    }
    __syncthreads();
  }

  if (tid < 16) {
    float mx = tv[tid * TOPK];                   // selections are descending
    float e[TOPK], ssum = 0.f;
#pragma unroll
    for (int j = 0; j < TOPK; ++j) { e[j] = __expf(tv[tid * TOPK + j] - mx); ssum += e[j]; }
    float inv = 1.f / ssum;
    size_t base = ((size_t)((c * GROUP + g) * NPROP) + row0 + tid) * TOPK;
#pragma unroll
    for (int j = 0; j < TOPK; ++j) {
      topv[base + j] = e[j] * inv;
      topi[base + j] = ti[tid * TOPK + j];
    }
  }
}

// ---------------------------------------------------------------------------
// K4: out[n,c,g*64+o] = conv_b + sum_j p_j * proj[c, idx_j, g*64+o]
//     (grouped conv commuted through the rank-20 attention sum)
// ---------------------------------------------------------------------------
__global__ void k_out(const float* __restrict__ topv, const int* __restrict__ topi,
                      const float* __restrict__ proj, const float* __restrict__ convb,
                      float* __restrict__ out) {
  int t = threadIdx.x;
  int o = t & 63, sub = t >> 6;
  int trip = blockIdx.x * 4 + sub;               // (c*16+g)*1024 + n
  int n = trip & 1023, g = (trip >> 10) & 15, c = trip >> 14;
  const float* p  = topv + (size_t)trip * TOPK;
  const int*   ix = topi + (size_t)trip * TOPK;
  float acc = convb[g * DG + o];
#pragma unroll
  for (int j = 0; j < TOPK; ++j)
    acc += p[j] * proj[((size_t)c * NPROP + ix[j]) * 1024 + g * DG + o];
  out[((size_t)n * CCLS + c) * 1024 + g * DG + o] = acc;
}

// ---------------------------------------------------------------------------
extern "C" void kernel_launch(void* const* d_in, const int* in_sizes, int n_in,
                              void* d_out, int out_size, void* d_ws, size_t ws_size,
                              hipStream_t stream) {
  (void)in_sizes; (void)n_in; (void)out_size; (void)ws_size;
  const float* f_a = (const float*)d_in[0];
  const float* pos = (const float*)d_in[1];
  const float* iou = (const float*)d_in[2];
  const float* WGw = (const float*)d_in[3];
  const float* WGb = (const float*)d_in[4];
  const float* WKw = (const float*)d_in[5];
  const float* WKb = (const float*)d_in[6];
  const float* WQw = (const float*)d_in[7];
  const float* WQb = (const float*)d_in[8];
  const float* CVw = (const float*)d_in[9];
  const float* CVb = (const float*)d_in[10];
  float* out = (float*)d_out;

  char* ws = (char*)d_ws;
  size_t off = 0;
  auto take = [&](size_t bytes) -> char* {
    char* p = ws + off;
    off = (off + bytes + 255) & ~(size_t)255;
    return p;
  };
  bf16*  fa_bf = (bf16*)take((size_t)CCLS * NPROP * FEATD * 2);          // 4 MB
  bf16*  wkT   = (bf16*)take((size_t)FEATD * GROUP * DG * 2);            // 2 MB
  bf16*  wqT   = (bf16*)take((size_t)FEATD * GROUP * DG * 2);            // 2 MB
  bf16*  wc    = (bf16*)take((size_t)GROUP * DG * FEATD * 2);            // 2 MB
  bf16*  Qb    = (bf16*)take((size_t)CCLS * NPROP * GROUP * DG * 2);     // 4 MB
  bf16*  Kb    = (bf16*)take((size_t)CCLS * NPROP * GROUP * DG * 2);     // 4 MB
  float* proj  = (float*)take((size_t)CCLS * NPROP * GROUP * DG * 4);    // 8 MB
  bf16*  gate  = (bf16*)take((size_t)CCLS * GROUP * NPROP * NPROP * 2);  // 64 MB
  float* topv  = (float*)take((size_t)CCLS * GROUP * NPROP * TOPK * 4);
  int*   topi  = (int*)take((size_t)CCLS * GROUP * NPROP * TOPK * 4);

  k_convert_A<<<8192, 256, 0, stream>>>(f_a, fa_bf);
  k_transpose_w<<<4096, 256, 0, stream>>>(WKw, wkT);
  k_transpose_w<<<4096, 256, 0, stream>>>(WQw, wqT);
  k_convert_w<<<4096, 256, 0, stream>>>(CVw, wc);
  k_gate<<<8192, 256, 0, stream>>>(pos, iou, WGw, WGb, gate);

  k_gemm<<<dim3(128, 2), 256, 0, stream>>>(fa_bf, wkT, WKb, Kb, nullptr);
  k_gemm<<<dim3(128, 2), 256, 0, stream>>>(fa_bf, wqT, WQb, Qb, nullptr);
  k_gemm<<<dim3(128, 2), 256, 0, stream>>>(fa_bf, wc, nullptr, nullptr, proj);

  size_t shmem = (size_t)16 * NPROP * 4 + 16 * TOPK * 4 + 16 * TOPK * 4 +
                 128 * 4 + 128 * 4;  // ~68 KB (gfx1250: 320 KB LDS per WGP)
  k_attn<<<dim3(NPROP / 16, GROUP, CCLS), 128, shmem, stream>>>(Qb, Kb, gate,
                                                                topv, topi);

  k_out<<<8192, 256, 0, stream>>>(topv, topi, proj, CVb, out);
}
